// DiffSOCSImager_1024x2048_33191507264137
// MI455X (gfx1250) — matrix operand order repaired and checked by reading, not compiled, not run on hardware
//
#include <hip/hip_runtime.h>
#include <math.h>

// ---------------------------------------------------------------------------
// DiffSOCS imager for MI455X (gfx1250, wave32).
//
// Algebraic collapse: M[p,k] = P1*conj(P2) = 1{f<=FC}1{f1<=FC}1{f2<=FC}
// (defocus phase cancels exactly), M is binary with support on ~171 pixels
// near DC.  SVD -> 64x64 Gram (EXACT in f16 WMMA: 0/1 entries, K<=192) +
// Jacobi eig + WMMA GEMM for U = M*Vg/s.  Kernels built by direct sparse
// inverse DFT on the 114x114 crop.  Heavy stage = 65 centered 2D FFTs of the
// 16MB array (radix-2 in LDS), ~5 GB traffic -> sub-ms at 23.3 TB/s; the
// whole working set is L2-resident (192 MB).
//
// CDNA5 paths: v_wmma_f32_16x16x32_f16 (Gram + U GEMMs),
// global_load_async_to_lds_b64 / global_store_async_from_lds_b64 +
// s_wait_asynccnt (FFT line staging, per-lane scattered LDS dest = fused
// ifftshift + bit-reversal with zero VGPR round-trip), global_prefetch_b8.
// ---------------------------------------------------------------------------

namespace {
constexpr int   H    = 1024;
constexpr int   W    = 2048;
constexpr int   HW   = H * W;
constexpr int   NSOCS = 32;
constexpr int   NSRC  = 64;
constexpr float FC   = 0.85f / 193.0f;          // NA / LAM
constexpr int   KSZ  = 114;                     // 2*HS, HS = 115//2
constexpr int   HS   = 57;
constexpr int   CH   = H / 2, CW = W / 2;
constexpr int   SUPX = 19, SUPY = 9;            // |kx|<=9, |ky|<=4
constexpr int   NSUP = SUPX * SUPY;             // 171
constexpr int   PADP = 192;                     // padded pixel rows (mult of 32)
constexpr float PI2  = 6.28318530717958647692f;
}

typedef __attribute__((ext_vector_type(16))) _Float16 v16h;
typedef __attribute__((ext_vector_type(8)))  float    v8f;

__device__ __forceinline__ float2 cmul(float2 a, float2 b) {
    return make_float2(a.x * b.x - a.y * b.y, a.x * b.y + a.y * b.x);
}

// ---------------------------------------------------------------------------
// CDNA5 async global<->LDS DMA (ASYNCcnt-tracked, per-lane LDS addresses).
// Generic LDS pointers are {aperture_hi32, lds_offset_lo32}; truncating to
// 32 bits yields the LDS byte offset expected in the VDST/VSRC VGPR.
// ---------------------------------------------------------------------------
__device__ __forceinline__ void async_load_f2_to_lds(const float2* gsrc, float2* ldst) {
    unsigned loff = (unsigned)(uintptr_t)ldst;
    asm volatile("global_load_async_to_lds_b64 %0, %1, off"
                 :: "v"(loff), "v"(gsrc) : "memory");
}
__device__ __forceinline__ void async_store_f2_from_lds(float2* gdst, const float2* lsrc) {
    unsigned loff = (unsigned)(uintptr_t)lsrc;
    asm volatile("global_store_async_from_lds_b64 %0, %1, off"
                 :: "v"(gdst), "v"(loff) : "memory");
}
__device__ __forceinline__ void wait_async0() {
    asm volatile("s_wait_asynccnt 0x0" ::: "memory");
}

// 16-bit A-matrix 16x32 (MxK) lane/element -> K mapping (ISA 05_wmma 7.12.2)
__device__ __forceinline__ int wmmaA_k(int lane, int t) {
    int v = t >> 1;
    return ((v & 4) ? 16 : 0) + ((lane & 16) ? 8 : 0) + ((v & 3) << 1) + (t & 1);
}
// 16-bit B-matrix 32x16 (KxN): n = lane&15, K = wmmaA_k(lane, t), same order as A

// ---------------------------------------------------------------------------
// Stage 0: source points fs[k] (needs sigma_c on device)
// ---------------------------------------------------------------------------
__global__ void k_source(const float* sigma_c, float* fs /* fsx[64], fsy[64] */) {
    int i = threadIdx.x;              // 64 threads
    float sc   = fminf(fmaxf(*sigma_c, 0.01f), 0.9f);
    float rmax = sc * FC;
    int   tI = i / 5, rI = i % 5;     // n_r = 5, n_theta = 13, xy-meshgrid ravel
    float r  = (0.25f * rI) * rmax;
    float th = tI * (PI2 / 12.0f);
    fs[i]      = r * cosf(th);
    fs[64 + i] = r * sinf(th);
}

// ---------------------------------------------------------------------------
// Stage 1: binary Msmall (PADP x 64) on the pupil support + support coords
// ---------------------------------------------------------------------------
__global__ void k_msmall(const float* fs, float* Ms, int* supK) {
    int p = threadIdx.x;              // 192 threads
    int ky = 0, kx = 0;
    if (p < NSUP) { ky = p / SUPX - 4; kx = p % SUPX - 9; }
    supK[p]        = ky;
    supK[PADP + p] = kx;
    float fy = (float)ky / (float)H;
    float fx = (float)kx / (float)W;
    bool inside = (hypotf(fx, fy) <= FC) && (p < NSUP);
    for (int k = 0; k < NSRC; ++k) {
        float hx = 0.5f * fs[k], hy = 0.5f * fs[64 + k];
        float f1 = hypotf(fx + hx, fy + hy);
        float f2 = hypotf(fx - hx, fy - hy);
        Ms[p * NSRC + k] = (inside && f1 <= FC && f2 <= FC) ? 1.0f : 0.0f;
    }
}

// ---------------------------------------------------------------------------
// Stage 2: G = Ms^T * Ms (64x64) via v_wmma_f32_16x16x32_f16.
// Exact: entries are 0/1, partial sums <= 192 < 2048 (f16 integer-exact),
// accumulation in f32. One block of 512 threads = 16 waves = 16 output tiles.
// ---------------------------------------------------------------------------
__global__ void k_gram_wmma(const float* Ms, float* G) {
    const int lane = threadIdx.x & 31;
    const int wave = threadIdx.x >> 5;       // 0..15
    const int ti = wave >> 2, tj = wave & 3; // 4x4 tiles of 16x16
    const int m  = lane & 15;
    v8f acc = {};
#pragma unroll
    for (int ch = 0; ch < PADP / 32; ++ch) {
        v16h a, b;
#pragma unroll
        for (int t = 0; t < 16; ++t) {
            int ka = ch * 32 + wmmaA_k(lane, t);            // A = Ms^T : A[m][k] = Ms[k][m]
            a[t] = (_Float16)Ms[ka * NSRC + (ti * 16 + m)];
            int kb = ch * 32 + wmmaA_k(lane, t);            // B = Ms   : B[k][n]
            b[t] = (_Float16)Ms[kb * NSRC + (tj * 16 + m)];
        }
        acc = __builtin_amdgcn_wmma_f32_16x16x32_f16(false, a, false, b,
                                                     (short)0, acc, false, false);
    }
    const int n = lane & 15;
    const int mb = (lane & 16) ? 8 : 0;
#pragma unroll
    for (int r = 0; r < 8; ++r)
        G[(ti * 16 + mb + r) * NSRC + (tj * 16 + n)] = acc[r];
}

// ---------------------------------------------------------------------------
// Stage 3: Jacobi eigensolve of symmetric 64x64 G; top-32 eigenpairs sorted.
// alpha = lambda (= s^2), invs = 1/sqrt(lambda), Vg32 = top-32 eigenvectors.
// ---------------------------------------------------------------------------
__global__ void k_jacobi(const float* G, float* alpha, float* invs, float* Vg) {
    __shared__ float A[64][64];
    __shared__ float V[64][64];
    __shared__ float shc, shs;
    const int t = threadIdx.x;   // 64 threads
    for (int j = 0; j < 64; ++j) {
        A[t][j] = G[t * 64 + j];
        V[t][j] = (t == j) ? 1.0f : 0.0f;
    }
    __syncthreads();
    for (int sweep = 0; sweep < 10; ++sweep) {
        for (int p = 0; p < 63; ++p) {
            for (int q = p + 1; q < 64; ++q) {
                if (t == 0) {
                    float apq = A[p][q];
                    if (fabsf(apq) < 1e-9f) { shc = 1.0f; shs = 0.0f; }
                    else {
                        float tau = (A[q][q] - A[p][p]) / (2.0f * apq);
                        float tt  = copysignf(1.0f, tau) /
                                    (fabsf(tau) + sqrtf(1.0f + tau * tau));
                        float c   = rsqrtf(1.0f + tt * tt);
                        shc = c; shs = tt * c;
                    }
                }
                __syncthreads();
                float c = shc, s = shs;
                float aip = A[t][p], aiq = A[t][q];
                A[t][p] = c * aip - s * aiq;
                A[t][q] = s * aip + c * aiq;
                float vip = V[t][p], viq = V[t][q];
                V[t][p] = c * vip - s * viq;
                V[t][q] = s * vip + c * viq;
                __syncthreads();
                float api = A[p][t], aqi = A[q][t];
                A[p][t] = c * api - s * aqi;
                A[q][t] = s * api + c * aqi;
                __syncthreads();
            }
        }
    }
    // Rank (descending, stable) and emit top-32.
    float li = A[t][t];
    int rank = 0;
    for (int j = 0; j < 64; ++j) {
        float lj = A[j][j];
        if (lj > li || (lj == li && j < t)) ++rank;
    }
    if (rank < NSOCS) {
        float lam = fmaxf(li, 0.0f);
        alpha[rank] = lam;
        float sv = sqrtf(lam);
        invs[rank] = (sv > 1e-12f) ? (1.0f / sv) : 0.0f;
        for (int r = 0; r < 64; ++r) Vg[r * NSOCS + rank] = V[r][t];
    }
}

// ---------------------------------------------------------------------------
// Stage 4: Usmall (PADP x 32) = Ms (PADP x 64) @ Vg (64 x 32), cols / s.
// 768 threads = 24 waves = 12x2 output tiles, WMMA f16 (Vg is O(1)).
// ---------------------------------------------------------------------------
__global__ void k_computeU_wmma(const float* Ms, const float* Vg,
                                const float* invs, float* U) {
    const int lane = threadIdx.x & 31;
    const int wave = threadIdx.x >> 5;        // 0..23
    const int mt = wave >> 1, nt = wave & 1;  // 12 x 2 tiles
    const int m  = lane & 15;
    v8f acc = {};
#pragma unroll
    for (int ch = 0; ch < 2; ++ch) {
        v16h a, b;
#pragma unroll
        for (int t = 0; t < 16; ++t) {
            int ka = ch * 32 + wmmaA_k(lane, t);            // A = Ms row-major
            a[t] = (_Float16)Ms[(mt * 16 + m) * NSRC + ka];
            int kb = ch * 32 + wmmaA_k(lane, t);            // B = Vg
            b[t] = (_Float16)Vg[kb * NSOCS + (nt * 16 + m)];
        }
        acc = __builtin_amdgcn_wmma_f32_16x16x32_f16(false, a, false, b,
                                                     (short)0, acc, false, false);
    }
    const int n  = nt * 16 + (lane & 15);
    const float sc = invs[n];
    const int mb = (lane & 16) ? 8 : 0;
#pragma unroll
    for (int r = 0; r < 8; ++r)
        U[(mt * 16 + mb + r) * NSOCS + n] = acc[r] * sc;
}

// ---------------------------------------------------------------------------
// Stage 5: kern[k] = center-114x114 crop of ifft2c(freq_k), direct sparse DFT
// over the <=171 spectral support points (centered transform).
// ---------------------------------------------------------------------------
__global__ void k_kern_direct(const float* U, const int* supK, float2* Kc) {
    int idx = blockIdx.x * blockDim.x + threadIdx.x;
    if (idx >= NSOCS * KSZ * KSZ) return;
    int k   = idx / (KSZ * KSZ);
    int rem = idx % (KSZ * KSZ);
    int yy  = rem / KSZ, xx = rem % KSZ;
    float ys = (float)(yy - HS) * (PI2 / (float)H);
    float xs = (float)(xx - HS) * (PI2 / (float)W);
    float re = 0.0f, im = 0.0f;
    for (int p = 0; p < PADP; ++p) {
        float u = U[p * NSOCS + k];
        float ang = (float)supK[p] * ys + (float)supK[PADP + p] * xs;
        float s, c;
        __sincosf(ang, &s, &c);
        re += u * c;
        im += u * s;
    }
    const float inv = 1.0f / ((float)H * (float)W);
    Kc[idx] = make_float2(re * inv, im * inv);
}

// ---------------------------------------------------------------------------
// Centered radix-2 FFT along a line (row or column), in LDS.
// Centering handled by (n + N/2) & (N-1) index remap (even N: fftshift ==
// ifftshift).  Load phase: async global->LDS DMA with per-lane scattered LDS
// destinations (fused ifftshift + bit-reversal, no VGPR round-trip).
// Forward store phase: async LDS->global DMA (scale == 1).
// ---------------------------------------------------------------------------
template <int LOGN, bool INV>
__global__ void k_fft_line(float2* data, int stride, int lineStride) {
    constexpr int N = 1 << LOGN;
    __shared__ float2 buf[N];
    const int tid = threadIdx.x;
    float2* base = data + (size_t)blockIdx.x * (size_t)lineStride;

    // Gather: LDS[bitrev(n)] = mem[ifftshift(n)] via ASYNCcnt-tracked DMA.
    for (int n = tid; n < N; n += blockDim.x) {          // uniform trip count
        int src = (n + N / 2) & (N - 1);                 // ifftshift
        int r   = (int)(__brev((unsigned)n) >> (32 - LOGN));
        async_load_f2_to_lds(base + (size_t)src * (size_t)stride, &buf[r]);
    }
    wait_async0();
    __syncthreads();

    for (int len = 2; len <= N; len <<= 1) {
        const int half = len >> 1;
        for (int b = tid; b < (N >> 1); b += blockDim.x) {
            int pos = b & (half - 1);
            int i0  = ((b ^ pos) << 1) | pos;
            int i1  = i0 + half;
            float ang = (INV ? PI2 : -PI2) * (float)pos / (float)len;
            float s, c;
            __sincosf(ang, &s, &c);
            float2 w = make_float2(c, s);
            float2 u = buf[i0];
            float2 v = cmul(w, buf[i1]);
            buf[i0] = make_float2(u.x + v.x, u.y + v.y);
            buf[i1] = make_float2(u.x - v.x, u.y - v.y);
        }
        __syncthreads();
    }

    if (!INV) {
        // Scale == 1: stream LDS -> global with async DMA stores.
        for (int k = tid; k < N; k += blockDim.x) {
            int dst = (k + N / 2) & (N - 1);             // fftshift
            async_store_f2_from_lds(base + (size_t)dst * (size_t)stride, &buf[k]);
        }
        wait_async0();
    } else {
        const float scale = 1.0f / (float)N;
        for (int k = tid; k < N; k += blockDim.x) {
            int dst = (k + N / 2) & (N - 1);
            float2 v = buf[k];
            base[(size_t)dst * (size_t)stride] = make_float2(v.x * scale, v.y * scale);
        }
    }
}

// ---------------------------------------------------------------------------
// Streaming helpers
// ---------------------------------------------------------------------------
__global__ void k_fill_u32(unsigned* p, unsigned v, size_t n) {
    for (size_t i = blockIdx.x * (size_t)blockDim.x + threadIdx.x; i < n;
         i += (size_t)gridDim.x * blockDim.x)
        p[i] = v;
}

__global__ void k_mask_to_c(const float* mask, float2* out) {
    int i = blockIdx.x * blockDim.x + threadIdx.x;
    if (i < HW) out[i] = make_float2(mask[i], 0.0f);
}

__global__ void k_scatter(const float2* Kc, float2* pad, int k) {
    int i = blockIdx.x * blockDim.x + threadIdx.x;
    if (i >= KSZ * KSZ) return;
    int yy = i / KSZ, xx = i % KSZ;
    pad[(size_t)(CH - HS + yy) * W + (CW - HS + xx)] = Kc[k * KSZ * KSZ + i];
}

__global__ void k_mulF(const float2* __restrict__ Fm, float2* __restrict__ buf) {
    int i = blockIdx.x * blockDim.x + threadIdx.x;
    if (i >= HW) return;
    __builtin_prefetch(&Fm[i + 8192], 0, 1);   // global_prefetch_b8, L2-resident stream
    buf[i] = cmul(Fm[i], buf[i]);
}

__global__ void k_accum(const float2* __restrict__ c, const float* alpha, int k,
                        float* __restrict__ I) {
    int i = blockIdx.x * blockDim.x + threadIdx.x;
    if (i >= HW) return;
    float2 v = c[i];
    I[i] += alpha[k] * (v.x * v.x + v.y * v.y);
}

__global__ void k_reduce_max(const float* I, unsigned* outmax) {
    __shared__ float sm[256];
    float m = 0.0f;
    for (size_t i = blockIdx.x * (size_t)blockDim.x + threadIdx.x; i < (size_t)HW;
         i += (size_t)gridDim.x * blockDim.x)
        m = fmaxf(m, I[i]);
    sm[threadIdx.x] = m;
    __syncthreads();
    for (int o = 128; o > 0; o >>= 1) {
        if ((int)threadIdx.x < o) sm[threadIdx.x] = fmaxf(sm[threadIdx.x], sm[threadIdx.x + o]);
        __syncthreads();
    }
    if (threadIdx.x == 0) atomicMax(outmax, __float_as_uint(sm[0]));  // I >= 0
}

__global__ void k_normalize(const float* I, const unsigned* maxu, float* out) {
    int i = blockIdx.x * blockDim.x + threadIdx.x;
    if (i >= HW) return;
    float mv = __uint_as_float(*maxu);
    out[i] = I[i] / mv;
}

// ---------------------------------------------------------------------------
// Host driver (graph-capture safe: kernels only, all on `stream`)
// ---------------------------------------------------------------------------
extern "C" void kernel_launch(void* const* d_in, const int* in_sizes, int n_in,
                              void* d_out, int out_size, void* d_ws, size_t ws_size,
                              hipStream_t stream) {
    (void)in_sizes; (void)n_in; (void)out_size; (void)ws_size;
    const float* d_mask  = (const float*)d_in[0];
    const float* d_sigma = (const float*)d_in[1];
    // d_in[2] = defocus_z4: cancels analytically in P1*conj(P2) -> unused.

    char* cur = (char*)d_ws;
    auto alloc = [&](size_t b) -> void* {
        void* p = (void*)cur;
        cur += (b + 255) & ~(size_t)255;
        return p;
    };
    float*    d_fs    = (float*)alloc(128 * sizeof(float));
    int*      d_sup   = (int*)alloc(2 * PADP * sizeof(int));
    float*    d_Ms    = (float*)alloc((size_t)PADP * NSRC * sizeof(float));
    float*    d_G     = (float*)alloc(64 * 64 * sizeof(float));
    float*    d_alpha = (float*)alloc(64 * sizeof(float));
    float*    d_invs  = (float*)alloc(64 * sizeof(float));
    float*    d_Vg    = (float*)alloc(64 * NSOCS * sizeof(float));
    float*    d_U     = (float*)alloc((size_t)PADP * NSOCS * sizeof(float));
    float2*   d_Kc    = (float2*)alloc((size_t)NSOCS * KSZ * KSZ * sizeof(float2));
    float2*   d_Fm    = (float2*)alloc((size_t)HW * sizeof(float2));
    float2*   d_pad   = (float2*)alloc((size_t)HW * sizeof(float2));
    float*    d_I     = (float*)alloc((size_t)HW * sizeof(float));
    unsigned* d_max   = (unsigned*)alloc(256);

    const int TPB  = 256;
    const int GHW  = (HW + TPB - 1) / TPB;

    // --- SOCS decomposition (tiny, WMMA-backed) ---
    k_source<<<1, 64, 0, stream>>>(d_sigma, d_fs);
    k_msmall<<<1, PADP, 0, stream>>>(d_fs, d_Ms, d_sup);
    k_gram_wmma<<<1, 512, 0, stream>>>(d_Ms, d_G);
    k_jacobi<<<1, 64, 0, stream>>>(d_G, d_alpha, d_invs, d_Vg);
    k_computeU_wmma<<<1, 768, 0, stream>>>(d_Ms, d_Vg, d_invs, d_U);
    k_kern_direct<<<(NSOCS * KSZ * KSZ + TPB - 1) / TPB, TPB, 0, stream>>>(d_U, d_sup, d_Kc);

    // --- Fm = fft2c(mask) ---
    k_mask_to_c<<<GHW, TPB, 0, stream>>>(d_mask, d_Fm);
    k_fft_line<11, false><<<H, TPB, 0, stream>>>(d_Fm, 1, W);   // rows, len 2048
    k_fft_line<10, false><<<W, TPB, 0, stream>>>(d_Fm, W, 1);   // cols, len 1024

    // --- Zero accumulators ---
    k_fill_u32<<<1024, TPB, 0, stream>>>((unsigned*)d_I, 0u, (size_t)HW);
    k_fill_u32<<<1, 64, 0, stream>>>(d_max, 0u, 1);

    // --- Per-SOCS-kernel imaging: c = ifft2c(Fm * fft2c(pad)); I += a|c|^2 ---
    for (int k = 0; k < NSOCS; ++k) {
        k_fill_u32<<<2048, TPB, 0, stream>>>((unsigned*)d_pad, 0u, (size_t)HW * 2);
        k_scatter<<<(KSZ * KSZ + TPB - 1) / TPB, TPB, 0, stream>>>(d_Kc, d_pad, k);
        k_fft_line<11, false><<<H, TPB, 0, stream>>>(d_pad, 1, W);
        k_fft_line<10, false><<<W, TPB, 0, stream>>>(d_pad, W, 1);
        k_mulF<<<GHW, TPB, 0, stream>>>(d_Fm, d_pad);
        k_fft_line<11, true><<<H, TPB, 0, stream>>>(d_pad, 1, W);
        k_fft_line<10, true><<<W, TPB, 0, stream>>>(d_pad, W, 1);
        k_accum<<<GHW, TPB, 0, stream>>>(d_pad, d_alpha, k, d_I);
    }

    // --- Normalize ---
    k_reduce_max<<<512, TPB, 0, stream>>>(d_I, d_max);
    k_normalize<<<GHW, TPB, 0, stream>>>(d_I, d_max, (float*)d_out);
}
